// RelativeMHSA_4690104287389
// MI455X (gfx1250) — matrix operand order repaired and checked by reading, not compiled, not run on hardware
//
#include <hip/hip_runtime.h>
#include <hip/hip_bf16.h>
#include <math.h>

// ---------------------------------------------------------------------------
// MI455X (gfx1250, wave32) Relative-position MHSA block.
// All GEMMs via v_wmma_f32_16x16x32_bf16 (bf16 operands, f32 accum).
// Double-buffered LDS staging so HBM->LDS overlaps WMMA instead of stalling
// on s_wait_dscnt 0 / barrier pairs every k-step.
// ---------------------------------------------------------------------------

typedef __attribute__((ext_vector_type(16))) __bf16 v16bf;
typedef __attribute__((ext_vector_type(8)))  float  v8f;

union ABFrag { v16bf v; uint4 q[2]; };

__device__ __forceinline__ unsigned short f2bf(float f) {
  unsigned u = __float_as_uint(f);
  u += 0x7FFFu + ((u >> 16) & 1u);          // round-to-nearest-even
  return (unsigned short)(u >> 16);
}
__device__ __forceinline__ float bf2f(unsigned short h) {
  return __uint_as_float(((unsigned)h) << 16);
}
__device__ __forceinline__ v8f zero8() {
  v8f z = {0.f,0.f,0.f,0.f,0.f,0.f,0.f,0.f};
  return z;
}

// ---------------------------------------------------------------------------
// Small prep kernels
// ---------------------------------------------------------------------------
__global__ void k_f32_to_bf16(const float* __restrict__ s,
                              unsigned short* __restrict__ d, int n) {
  for (int i = blockIdx.x * blockDim.x + threadIdx.x; i < n;
       i += gridDim.x * blockDim.x)
    d[i] = f2bf(s[i]);
}

// dst[o*K + k] = src[k*Nout + o]  (weights stored (in,out) -> (out,in) bf16)
__global__ void k_transpose_bf16(const float* __restrict__ s,
                                 unsigned short* __restrict__ d,
                                 int K, int Nout) {
  int n = K * Nout;
  for (int i = blockIdx.x * blockDim.x + threadIdx.x; i < n;
       i += gridDim.x * blockDim.x) {
    int o = i / K, k = i - o * K;
    d[i] = f2bf(s[(size_t)k * Nout + o]);
  }
}

// bias_full[h][n][m] = bias_table[rel_index[n][m]][h]   (12 x 1024 x 1024 bf16)
__global__ void k_bias_expand(const float* __restrict__ table,
                              const int* __restrict__ relidx,
                              unsigned short* __restrict__ d) {
  const int NN = 1024 * 1024;
  for (int i = blockIdx.x * blockDim.x + threadIdx.x; i < 12 * NN;
       i += gridDim.x * blockDim.x) {
    int h = i / NN, rem = i - h * NN;          // rem = n*1024 + m
    d[i] = f2bf(table[relidx[rem] * 12 + h]);
  }
}

// ---------------------------------------------------------------------------
// QKV GEMM: (16384 x 768) @ (768 x 2304) + bias, scattered to per-head
// q (B,H,N,64), k (B,H,N,64), v^T (B,H,64,N), all bf16.
// Tile 128x64, BK=32, 256 threads = 8 waves in a 4x2 grid of 32x32 patches.
// Double-buffered LDS: one barrier per k-step.
// ---------------------------------------------------------------------------
__global__ __launch_bounds__(256) void k_gemm_qkv(
    const unsigned short* __restrict__ xb,   // (16384,768) bf16 row-major
    const unsigned short* __restrict__ wT,   // (2304,768)  bf16 row-major
    const float* __restrict__ bias,          // (2304)
    unsigned short* __restrict__ qg,
    unsigned short* __restrict__ kg,
    unsigned short* __restrict__ vTg) {
  __shared__ unsigned short At[2][128][32];
  __shared__ unsigned short Bt[2][64][32];
  const int t = threadIdx.x;
  const int m0 = blockIdx.y * 128;
  const int o0 = blockIdx.x * 64;
  const int w = t >> 5, lane = t & 31, l16 = lane & 15, hi = lane >> 4;
  const int wm = w >> 1, wn = w & 1;

  const int arow = t >> 1, akk = (t & 1) * 16;     // A staging map (32B/thr)
  const int brow = t >> 2, bkk = (t & 3) * 8;      // B staging map (16B/thr)

  v8f c[2][2];
  for (int i = 0; i < 2; ++i)
    for (int j = 0; j < 2; ++j) c[i][j] = zero8();

  // prologue: stage k0 = 0 into buffer 0
  {
    const uint4* s = (const uint4*)(xb + (size_t)(m0 + arow) * 768 + akk);
    uint4* d = (uint4*)&At[0][arow][akk];
    d[0] = s[0]; d[1] = s[1];
    *(uint4*)&Bt[0][brow][bkk] =
        *(const uint4*)(wT + (size_t)(o0 + brow) * 768 + bkk);
  }
  __syncthreads();

  for (int kt = 0; kt < 24; ++kt) {
    const int cur = kt & 1;
    if (kt + 1 < 24) {  // stage next tile into the other buffer
      const int k0n = (kt + 1) * 32;
      const uint4* s =
          (const uint4*)(xb + (size_t)(m0 + arow) * 768 + k0n + akk);
      uint4* d = (uint4*)&At[cur ^ 1][arow][akk];
      d[0] = s[0]; d[1] = s[1];
      *(uint4*)&Bt[cur ^ 1][brow][bkk] =
          *(const uint4*)(wT + (size_t)(o0 + brow) * 768 + k0n + bkk);
    }
    ABFrag a[2], b[2];
    for (int i = 0; i < 2; ++i) {  // A frag: lane=row%16, K chunks by lane/16
      const uint4* p = (const uint4*)&At[cur][wm * 32 + i * 16 + l16][0];
      a[i].q[0] = p[2 * hi]; a[i].q[1] = p[2 * hi + 1];
    }
    for (int j = 0; j < 2; ++j) {  // B frag: lane=col, 16 contiguous K
      const uint4* p = (const uint4*)&Bt[cur][wn * 32 + j * 16 + l16][0];
      b[j].q[0] = p[2 * hi]; b[j].q[1] = p[2 * hi + 1];
    }
    for (int i = 0; i < 2; ++i)
      for (int j = 0; j < 2; ++j)
        c[i][j] = __builtin_amdgcn_wmma_f32_16x16x32_bf16(
            false, a[i].v, false, b[j].v, (short)0, c[i][j], false, false);
    __syncthreads();
  }

  // Epilogue: +bias, split q/k/v, per-head layouts
  for (int i = 0; i < 2; ++i)
    for (int j = 0; j < 2; ++j)
      for (int r = 0; r < 8; ++r) {
        int gm = m0 + wm * 32 + i * 16 + r + 8 * hi;  // C row = r + 8*(lane/16)
        int o  = o0 + wn * 32 + j * 16 + l16;         // C col = lane%16
        float val = c[i][j][r] + bias[o];
        int which = o / 768, rem = o - which * 768;
        int h = rem >> 6, d = rem & 63;
        int bi = gm >> 10, n = gm & 1023;
        unsigned short hv = f2bf(val);
        size_t bh = (size_t)bi * 12 + h;
        if (which == 0)      qg[(bh * 1024 + n) * 64 + d] = hv;
        else if (which == 1) kg[(bh * 1024 + n) * 64 + d] = hv;
        else                 vTg[(bh * 64 + d) * 1024 + n] = hv;
      }
}

// ---------------------------------------------------------------------------
// Flash attention per (b,h): grid (16 q-tiles, 192 bh). 128 threads = 4 waves,
// each wave owns 16 query rows. Online softmax, WMMA for QK^T and PV.
// K/V/bias tiles double-buffered in LDS.
// ---------------------------------------------------------------------------
__global__ __launch_bounds__(128) void k_flash(
    const unsigned short* __restrict__ qg,     // (B,H,N,64)
    const unsigned short* __restrict__ kg,     // (B,H,N,64)
    const unsigned short* __restrict__ vTg,    // (B,H,64,N)
    const unsigned short* __restrict__ biasF,  // (H,N,N) bf16
    unsigned short* __restrict__ og) {         // (B,N,H,64) bf16
  __shared__ unsigned short Kt[2][64][64];
  __shared__ unsigned short Vt[2][64][64];
  __shared__ unsigned short Bb[2][64][64];
  __shared__ unsigned short Pt[4][16][64];
  const int bh = blockIdx.y;
  const int bi = bh / 12, h = bh % 12;
  const int q0 = blockIdx.x * 64;
  const int t = threadIdx.x, w = t >> 5, lane = t & 31;
  const int l16 = lane & 15, hi = lane >> 4;

  // q tile held as A fragments in registers (hd=64 -> two K=32 frags)
  ABFrag qa[2];
  {
    const uint4* p =
        (const uint4*)(qg + ((size_t)bh * 1024 + q0 + w * 16 + l16) * 64);
    qa[0].q[0] = p[2 * hi];     qa[0].q[1] = p[2 * hi + 1];
    qa[1].q[0] = p[4 + 2 * hi]; qa[1].q[1] = p[4 + 2 * hi + 1];
  }

  float mrun[8], lrun[8];
  v8f oacc[4];
  for (int r = 0; r < 8; ++r) { mrun[r] = -INFINITY; lrun[r] = 0.f; }
  for (int j = 0; j < 4; ++j) oacc[j] = zero8();

  const int srow = t >> 1, scol = (t & 1) * 32;  // cooperative staging map
  const unsigned short* kbase = kg + (size_t)bh * 1024 * 64 + (size_t)srow * 64 + scol;
  const unsigned short* vbase = vTg + ((size_t)bh * 64 + srow) * 1024 + scol;
  const unsigned short* bbase =
      biasF + ((size_t)h * 1024 + q0 + srow) * 1024 + scol;

  // prologue: stage kv block 0 into buffer 0
  {
    const uint4* s0 = (const uint4*)kbase;
    uint4* d0 = (uint4*)&Kt[0][srow][scol];
    d0[0] = s0[0]; d0[1] = s0[1]; d0[2] = s0[2]; d0[3] = s0[3];
    const uint4* s1 = (const uint4*)vbase;
    uint4* d1 = (uint4*)&Vt[0][srow][scol];
    d1[0] = s1[0]; d1[1] = s1[1]; d1[2] = s1[2]; d1[3] = s1[3];
    const uint4* s2 = (const uint4*)bbase;
    uint4* d2 = (uint4*)&Bb[0][srow][scol];
    d2[0] = s2[0]; d2[1] = s2[1]; d2[2] = s2[2]; d2[3] = s2[3];
  }
  __syncthreads();

  for (int it = 0; it < 16; ++it) {
    const int cur = it & 1;
    if (it + 1 < 16) {  // stage next KV block into the other buffer
      const int n1 = (it + 1) * 64;
      const uint4* s0 = (const uint4*)(kbase + (size_t)n1 * 64);
      uint4* d0 = (uint4*)&Kt[cur ^ 1][srow][scol];
      d0[0] = s0[0]; d0[1] = s0[1]; d0[2] = s0[2]; d0[3] = s0[3];
      const uint4* s1 = (const uint4*)(vbase + n1);
      uint4* d1 = (uint4*)&Vt[cur ^ 1][srow][scol];
      d1[0] = s1[0]; d1[1] = s1[1]; d1[2] = s1[2]; d1[3] = s1[3];
      const uint4* s2 = (const uint4*)(bbase + n1);
      uint4* d2 = (uint4*)&Bb[cur ^ 1][srow][scol];
      d2[0] = s2[0]; d2[1] = s2[1]; d2[2] = s2[2]; d2[3] = s2[3];
      if (it + 2 < 16) {  // lowers to global_prefetch_b8 (tile after next)
        __builtin_prefetch(kbase + (size_t)(n1 + 64) * 64, 0, 0);
        __builtin_prefetch(vbase + n1 + 64, 0, 0);
      }
    }

    // S = Q K^T (16 x 64 strip per wave, 4 col subtiles x 2 K-halves)
    v8f s4[4];
    for (int j = 0; j < 4; ++j) {
      ABFrag b0, b1;
      const uint4* p = (const uint4*)&Kt[cur][j * 16 + l16][0];
      b0.q[0] = p[2 * hi];     b0.q[1] = p[2 * hi + 1];
      b1.q[0] = p[4 + 2 * hi]; b1.q[1] = p[4 + 2 * hi + 1];
      v8f sj = zero8();
      sj = __builtin_amdgcn_wmma_f32_16x16x32_bf16(false, qa[0].v, false, b0.v,
                                                   (short)0, sj, false, false);
      sj = __builtin_amdgcn_wmma_f32_16x16x32_bf16(false, qa[1].v, false, b1.v,
                                                   (short)0, sj, false, false);
      s4[j] = sj;
    }

    // scale + relative-position bias
    for (int j = 0; j < 4; ++j)
      for (int r = 0; r < 8; ++r)
        s4[j][r] = s4[j][r] * 0.125f +
                   bf2f(Bb[cur][w * 16 + r + 8 * hi][j * 16 + l16]);

    // online softmax update (row reductions across the 16-lane half)
    for (int r = 0; r < 8; ++r) {
      float m = s4[0][r];
      for (int j = 1; j < 4; ++j) m = fmaxf(m, s4[j][r]);
      for (int msk = 1; msk < 16; msk <<= 1)
        m = fmaxf(m, __shfl_xor(m, msk, 32));
      float mnew = fmaxf(mrun[r], m);
      float alpha = __expf(mrun[r] - mnew);
      float rsum = 0.f;
      for (int j = 0; j < 4; ++j) {
        float pv = __expf(s4[j][r] - mnew);
        s4[j][r] = pv;
        rsum += pv;
      }
      for (int msk = 1; msk < 16; msk <<= 1)
        rsum += __shfl_xor(rsum, msk, 32);
      lrun[r] = lrun[r] * alpha + rsum;
      mrun[r] = mnew;
      for (int j = 0; j < 4; ++j) oacc[j][r] *= alpha;
    }

    // relayout P (C-layout) -> A-fragment layout via per-wave LDS
    for (int j = 0; j < 4; ++j)
      for (int r = 0; r < 8; ++r)
        Pt[w][r + 8 * hi][j * 16 + l16] = f2bf(s4[j][r]);
    // same-wave DS ops are in-order: safe to read back without a barrier
    ABFrag pa[2];
    {
      const uint4* p = (const uint4*)&Pt[w][l16][0];
      pa[0].q[0] = p[2 * hi];     pa[0].q[1] = p[2 * hi + 1];
      pa[1].q[0] = p[4 + 2 * hi]; pa[1].q[1] = p[4 + 2 * hi + 1];
    }

    // O += P V  (K = 64 keys -> 2 WMMA per d-subtile)
    for (int j = 0; j < 4; ++j) {
      ABFrag vb0, vb1;
      const uint4* p = (const uint4*)&Vt[cur][j * 16 + l16][0];
      vb0.q[0] = p[2 * hi];     vb0.q[1] = p[2 * hi + 1];
      vb1.q[0] = p[4 + 2 * hi]; vb1.q[1] = p[4 + 2 * hi + 1];
      oacc[j] = __builtin_amdgcn_wmma_f32_16x16x32_bf16(
          false, pa[0].v, false, vb0.v, (short)0, oacc[j], false, false);
      oacc[j] = __builtin_amdgcn_wmma_f32_16x16x32_bf16(
          false, pa[1].v, false, vb1.v, (short)0, oacc[j], false, false);
    }
    __syncthreads();
  }

  // normalize and write (B,N,H,64) bf16
  for (int r = 0; r < 8; ++r) {
    float inv = 1.0f / lrun[r];
    int grow = q0 + w * 16 + r + 8 * hi;
    for (int j = 0; j < 4; ++j) {
      int d = j * 16 + l16;
      og[(((size_t)bi * 1024 + grow) * 12 + h) * 64 + d] =
          f2bf(oacc[j][r] * inv);
    }
  }
}

// ---------------------------------------------------------------------------
// Proj GEMM fused with bias + residual: y = attn_out @ W + b + x  (f32 out)
// ---------------------------------------------------------------------------
__global__ __launch_bounds__(256) void k_gemm_proj(
    const unsigned short* __restrict__ ab,   // (16384,768) bf16
    const unsigned short* __restrict__ wT,   // (768,768)  bf16
    const float* __restrict__ bias,
    const float* __restrict__ xres,
    float* __restrict__ y) {
  __shared__ unsigned short At[2][128][32];
  __shared__ unsigned short Bt[2][64][32];
  const int t = threadIdx.x;
  const int m0 = blockIdx.y * 128;
  const int o0 = blockIdx.x * 64;
  const int w = t >> 5, lane = t & 31, l16 = lane & 15, hi = lane >> 4;
  const int wm = w >> 1, wn = w & 1;

  const int arow = t >> 1, akk = (t & 1) * 16;
  const int brow = t >> 2, bkk = (t & 3) * 8;

  v8f c[2][2];
  for (int i = 0; i < 2; ++i)
    for (int j = 0; j < 2; ++j) c[i][j] = zero8();

  {
    const uint4* s = (const uint4*)(ab + (size_t)(m0 + arow) * 768 + akk);
    uint4* d = (uint4*)&At[0][arow][akk];
    d[0] = s[0]; d[1] = s[1];
    *(uint4*)&Bt[0][brow][bkk] =
        *(const uint4*)(wT + (size_t)(o0 + brow) * 768 + bkk);
  }
  __syncthreads();

  for (int kt = 0; kt < 24; ++kt) {
    const int cur = kt & 1;
    if (kt + 1 < 24) {
      const int k0n = (kt + 1) * 32;
      const uint4* s =
          (const uint4*)(ab + (size_t)(m0 + arow) * 768 + k0n + akk);
      uint4* d = (uint4*)&At[cur ^ 1][arow][akk];
      d[0] = s[0]; d[1] = s[1];
      *(uint4*)&Bt[cur ^ 1][brow][bkk] =
          *(const uint4*)(wT + (size_t)(o0 + brow) * 768 + k0n + bkk);
    }
    ABFrag a[2], b[2];
    for (int i = 0; i < 2; ++i) {
      const uint4* p = (const uint4*)&At[cur][wm * 32 + i * 16 + l16][0];
      a[i].q[0] = p[2 * hi]; a[i].q[1] = p[2 * hi + 1];
    }
    for (int j = 0; j < 2; ++j) {
      const uint4* p = (const uint4*)&Bt[cur][wn * 32 + j * 16 + l16][0];
      b[j].q[0] = p[2 * hi]; b[j].q[1] = p[2 * hi + 1];
    }
    for (int i = 0; i < 2; ++i)
      for (int j = 0; j < 2; ++j)
        c[i][j] = __builtin_amdgcn_wmma_f32_16x16x32_bf16(
            false, a[i].v, false, b[j].v, (short)0, c[i][j], false, false);
    __syncthreads();
  }

  for (int i = 0; i < 2; ++i)
    for (int j = 0; j < 2; ++j)
      for (int r = 0; r < 8; ++r) {
        int gm = m0 + wm * 32 + i * 16 + r + 8 * hi;
        int o  = o0 + wn * 32 + j * 16 + l16;
        size_t idx = (size_t)gm * 768 + o;
        y[idx] = c[i][j][r] + bias[o] + xres[idx];
      }
}

// ---------------------------------------------------------------------------
// In-place LayerNorm over D=768 per row (one block per row, 256 threads).
// ---------------------------------------------------------------------------
__global__ __launch_bounds__(256) void k_ln(float* __restrict__ y,
                                            const float* __restrict__ g,
                                            const float* __restrict__ b) {
  int row = blockIdx.x, t = threadIdx.x;
  float* p = y + (size_t)row * 768;
  float v0 = p[t], v1 = p[t + 256], v2 = p[t + 512];
  float s = v0 + v1 + v2, ss = v0 * v0 + v1 * v1 + v2 * v2;
  for (int m = 16; m >= 1; m >>= 1) {
    s += __shfl_xor(s, m, 32);
    ss += __shfl_xor(ss, m, 32);
  }
  __shared__ float a1[8], a2[8];
  if ((t & 31) == 0) { a1[t >> 5] = s; a2[t >> 5] = ss; }
  __syncthreads();
  float ts = 0.f, tss = 0.f;
  for (int i = 0; i < 8; ++i) { ts += a1[i]; tss += a2[i]; }
  float mu = ts * (1.0f / 768.0f);
  float var = tss * (1.0f / 768.0f) - mu * mu;
  float rs = rsqrtf(var + 1e-5f);
  p[t]       = (v0 - mu) * rs * g[t]       + b[t];
  p[t + 256] = (v1 - mu) * rs * g[t + 256] + b[t + 256];
  p[t + 512] = (v2 - mu) * rs * g[t + 512] + b[t + 512];
}

// ---------------------------------------------------------------------------
extern "C" void kernel_launch(void* const* d_in, const int* in_sizes, int n_in,
                              void* d_out, int out_size, void* d_ws,
                              size_t ws_size, hipStream_t stream) {
  (void)in_sizes; (void)n_in; (void)out_size; (void)ws_size;
  const float* x      = (const float*)d_in[0];
  const float* qkv_w  = (const float*)d_in[1];
  const float* qkv_b  = (const float*)d_in[2];
  const float* proj_w = (const float*)d_in[3];
  const float* proj_b = (const float*)d_in[4];
  const float* norm_g = (const float*)d_in[5];
  const float* norm_b = (const float*)d_in[6];
  const float* btab   = (const float*)d_in[7];
  const int*   ridx   = (const int*)d_in[8];
  float* y = (float*)d_out;

  char* ws = (char*)d_ws;
  size_t off = 0;
  auto take = [&](size_t bytes) -> void* {
    void* p = ws + off;
    off += (bytes + 255) & ~(size_t)255;
    return p;
  };
  unsigned short* xb    = (unsigned short*)take((size_t)16384 * 768 * 2);
  unsigned short* wqT   = (unsigned short*)take((size_t)2304 * 768 * 2);
  unsigned short* wpT   = (unsigned short*)take((size_t)768 * 768 * 2);
  unsigned short* qg    = (unsigned short*)take((size_t)16 * 12 * 1024 * 64 * 2);
  unsigned short* kg    = (unsigned short*)take((size_t)16 * 12 * 1024 * 64 * 2);
  unsigned short* vTg   = (unsigned short*)take((size_t)16 * 12 * 1024 * 64 * 2);
  unsigned short* biasF = (unsigned short*)take((size_t)12 * 1024 * 1024 * 2);
  unsigned short* ao    = (unsigned short*)take((size_t)16384 * 768 * 2);

  k_f32_to_bf16<<<4096, 256, 0, stream>>>(x, xb, 16384 * 768);
  k_transpose_bf16<<<4096, 256, 0, stream>>>(qkv_w, wqT, 768, 2304);
  k_transpose_bf16<<<2048, 256, 0, stream>>>(proj_w, wpT, 768, 768);
  k_bias_expand<<<8192, 256, 0, stream>>>(btab, ridx, biasF);
  k_gemm_qkv<<<dim3(36, 128), 256, 0, stream>>>(xb, wqT, qkv_b, qg, kg, vTg);
  k_flash<<<dim3(16, 192), 128, 0, stream>>>(qg, kg, vTg, biasF, ao);
  k_gemm_proj<<<dim3(12, 128), 256, 0, stream>>>(ao, wpT, proj_b, x, y);
  k_ln<<<16384, 256, 0, stream>>>(y, norm_g, norm_b);
}
